// SchmidhuberLinear_32255204393227
// MI455X (gfx1250) — matrix operand order repaired and checked by reading, not compiled, not run on hardware
//
#include <hip/hip_runtime.h>

typedef __attribute__((ext_vector_type(16))) _Float16 v16h;
typedef __attribute__((ext_vector_type(8)))  _Float16 v8h;
typedef __attribute__((ext_vector_type(8)))  float    v8f;

#define B_  512
#define D_  1024
#define H1_ 256
#define H2_ 128

// LDS layout (bytes):
//   [0, 262144)       per-wave h buffers: wave w owns [w*8192, w*8192+8192)
//                     phase1: 16x256 f16 (h1 strip), phase2: 16x128 f32 (h2 strip)
//   [262144, 270336)  stage: B-fragment-swizzled f16 weight tile (32x128 = 8KB)
//   [270336, 270848)  ssum[128] f32
//   [270848, 271360)  ssq[128]  f32
#define SMEM_BYTES 271360

extern "C" __global__ __launch_bounds__(1024, 1)
void loo_mlp_kernel(const float* __restrict__ x,
                    const float* __restrict__ W1,
                    const float* __restrict__ g1,
                    const float* __restrict__ be1,
                    const float* __restrict__ W2,
                    const float* __restrict__ g2,
                    const float* __restrict__ be2,
                    const float* __restrict__ W3,
                    const float* __restrict__ b3,
                    float* __restrict__ out)
{
    extern __shared__ char smem[];
    _Float16* hbuf_h = (_Float16*)smem;              // phase1 h1 (f16)
    float*    hbuf_f = (float*)smem;                 // phase2 h2 (f32)
    _Float16* stage  = (_Float16*)(smem + 262144);
    float*    ssum   = (float*)(smem + 270336);
    float*    ssq    = (float*)(smem + 270848);

    const int tid  = threadIdx.x;
    const int lane = tid & 31;
    const int w    = tid >> 5;        // wave id 0..31, owns batch rows [16w,16w+16)
    const int k    = blockIdx.x;      // predictor index 0..1023
    const int ln16 = lane & 15;
    const int half = lane >> 4;       // A-frag lane-group (K split per ISA layout)
    const int mrow = w * 16 + ln16;   // this lane's batch row in the A fragment
    const float inv_n = 1.0f / (float)B_;

    if (tid < H2_) { ssum[tid] = 0.f; ssq[tid] = 0.f; }

    // ------------- Phase 1: h1 = x @ W1[k]   (512 x 1024 x 256) -------------
    // Two N-half passes (cols [0,128) then [128,256)) keep accumulators at
    // 64 VGPRs/lane (8 tiles) to avoid spills. W1 is still read exactly once.
    for (int nh = 0; nh < 2; ++nh) {
        const int ncol0 = nh * H2_;                  // 0 or 128
        v8f acc1[8] = {};

        for (int kk = 0; kk < D_; kk += 32) {
            __syncthreads();
            // Stage W1[k, kk:kk+32, ncol0:ncol0+128] as f16, swizzled into
            // B-fragment order: stage[t*512 + lane*16 + kidx]
            {
                const int e0 = tid * 4;              // 4096 elements / 1024 threads
                const int r  = e0 >> 7;              // K row 0..31
                const int c  = e0 & 127;             // local N col (multiple of 4)
                float4 f0 = *(const float4*)
                    &W1[(((size_t)k * D_) + (size_t)(kk + r)) * H1_ + ncol0 + c];
                float fv[4] = {f0.x, f0.y, f0.z, f0.w};
                const int lane_ = ((r >> 3) & 1) * 16;
                const int kidx  = (r & 7) + ((r >> 4) << 3);
                #pragma unroll
                for (int i = 0; i < 4; ++i) {
                    int cc = c + i;
                    int t_ = cc >> 4, n_ = cc & 15;
                    stage[t_ * 512 + (lane_ + n_) * 16 + kidx] = (_Float16)fv[i];
                }
            }
            __syncthreads();
            // A fragment: row mrow; per ISA 16-bit A layout:
            // lanes 0-15: K=kk+[0..7],kk+[16..23]; lanes 16-31: kk+[8..15],kk+[24..31]
            v16h a;
            {
                const float*  xr = &x[(size_t)mrow * D_ + kk + half * 8];
                const float4* p  = (const float4*)xr;
                float4 q0 = p[0], q1 = p[1];
                const float4* p2 = (const float4*)(xr + 16);
                float4 q2 = p2[0], q3 = p2[1];
                float qa[16] = {q0.x,q0.y,q0.z,q0.w, q1.x,q1.y,q1.z,q1.w,
                                q2.x,q2.y,q2.z,q2.w, q3.x,q3.y,q3.z,q3.w};
                #pragma unroll
                for (int i = 0; i < 16; ++i) a[i] = (_Float16)qa[i];
            }
            #pragma unroll
            for (int t = 0; t < 8; ++t) {
                const v16h* bp = (const v16h*)&stage[t * 512 + lane * 16];
                acc1[t] = __builtin_amdgcn_wmma_f32_16x16x32_f16(
                    false, a, false, *bp, (short)0, acc1[t], false, false);
            }
        }

        // --------- BatchNorm1 + ReLU for this column half (b1 cancels) -------
        #pragma unroll
        for (int t = 0; t < 8; ++t) {
            float s = 0.f, q = 0.f;
            #pragma unroll
            for (int j = 0; j < 8; ++j) { float v = acc1[t][j]; s += v; q += v * v; }
            int c = t * 16 + ln16;                   // local col 0..127
            atomicAdd(&ssum[c], s);
            atomicAdd(&ssq[c], q);
        }
        __syncthreads();
        #pragma unroll
        for (int t = 0; t < 8; ++t) {
            int   cl  = t * 16 + ln16;
            int   c   = ncol0 + cl;
            float mu  = ssum[cl] * inv_n;
            float var = ssq[cl] * inv_n - mu * mu;
            float gs  = g1[(size_t)k * H1_ + c] * rsqrtf(var + 1e-5f);
            float be  = be1[(size_t)k * H1_ + c];
            #pragma unroll
            for (int j = 0; j < 8; ++j) {
                float v = (acc1[t][j] - mu) * gs + be;
                v = v > 0.f ? v : 0.f;
                int row = half * 8 + j;              // C layout -> strip row
                hbuf_h[w * 4096 + row * 256 + c] = (_Float16)v;
            }
        }
        __syncthreads();
        if (tid < H2_) { ssum[tid] = 0.f; ssq[tid] = 0.f; }   // reset stats
    }

    // ------------- Phase 2: h2 = h1 @ W2[k]  (512 x 256 x 128) ---------------
    v8f acc2[8] = {};
    for (int kk = 0; kk < H1_; kk += 32) {
        __syncthreads();
        // Stage W2[k, kk:kk+32, 0:128] swizzled into B-fragment order
        {
            const int e0 = tid * 4;                  // 4096 elements / 1024 threads
            const int r  = e0 >> 7;                  // K row 0..31
            const int c  = e0 & 127;                 // N col (multiple of 4)
            float4 f0 = *(const float4*)
                &W2[(((size_t)k * H1_) + (size_t)(kk + r)) * H2_ + c];
            float fv[4] = {f0.x, f0.y, f0.z, f0.w};
            const int lane_ = ((r >> 3) & 1) * 16;
            const int kidx  = (r & 7) + ((r >> 4) << 3);
            #pragma unroll
            for (int i = 0; i < 4; ++i) {
                int cc = c + i;
                int t_ = cc >> 4, n_ = cc & 15;
                stage[t_ * 512 + (lane_ + n_) * 16 + kidx] = (_Float16)fv[i];
            }
        }
        __syncthreads();
        // A fragment from this wave's own h1 strip (row-major 16x256 f16 in LDS)
        v16h a;
        {
            const _Float16* hp = &hbuf_h[w * 4096 + ln16 * 256 + kk + half * 8];
            v8h lo = *(const v8h*)hp;
            v8h hi = *(const v8h*)(hp + 16);
            #pragma unroll
            for (int i = 0; i < 8; ++i) { a[i] = lo[i]; a[8 + i] = hi[i]; }
        }
        #pragma unroll
        for (int t = 0; t < 8; ++t) {
            const v16h* bp = (const v16h*)&stage[t * 512 + lane * 16];
            acc2[t] = __builtin_amdgcn_wmma_f32_16x16x32_f16(
                false, a, false, *bp, (short)0, acc2[t], false, false);
        }
    }

    // ------------- BatchNorm2 + ReLU (b2 cancels) ----------------------------
    #pragma unroll
    for (int t = 0; t < 8; ++t) {
        float s = 0.f, q = 0.f;
        #pragma unroll
        for (int j = 0; j < 8; ++j) { float v = acc2[t][j]; s += v; q += v * v; }
        int c = t * 16 + ln16;
        atomicAdd(&ssum[c], s);
        atomicAdd(&ssq[c], q);
    }
    __syncthreads();
    #pragma unroll
    for (int t = 0; t < 8; ++t) {
        int   c   = t * 16 + ln16;
        float mu  = ssum[c] * inv_n;
        float var = ssq[c] * inv_n - mu * mu;
        float gs  = g2[(size_t)k * H2_ + c] * rsqrtf(var + 1e-5f);
        float be  = be2[(size_t)k * H2_ + c];
        #pragma unroll
        for (int j = 0; j < 8; ++j) {
            float v = (acc2[t][j] - mu) * gs + be;
            v = v > 0.f ? v : 0.f;
            int row = half * 8 + j;
            hbuf_f[w * 2048 + row * 128 + c] = v;    // h2 strip, f32 (reuses h1 space)
        }
    }
    __syncthreads();

    // ------------- y[b,k] = h2[b,:] . W3[k,:] + b3[k] ------------------------
    if (tid < B_) {
        const int b = tid;
        const float4* hr = (const float4*)&hbuf_f[b * H2_];   // == row b of h2
        float s = 0.f;
        #pragma unroll 8
        for (int j = 0; j < H2_ / 4; ++j) {
            float4 h4 = hr[j];
            float4 w4 = *(const float4*)&W3[(size_t)k * H2_ + j * 4];
            s += h4.x * w4.x + h4.y * w4.y + h4.z * w4.z + h4.w * w4.w;
        }
        out[(size_t)b * D_ + k] = s + b3[k];
    }
}

extern "C" void kernel_launch(void* const* d_in, const int* in_sizes, int n_in,
                              void* d_out, int out_size, void* d_ws, size_t ws_size,
                              hipStream_t stream) {
    const float* x   = (const float*)d_in[0];
    const float* W1  = (const float*)d_in[1];
    // d_in[2] = b1 : added before BN -> cancels exactly, unused
    const float* g1  = (const float*)d_in[3];
    const float* be1 = (const float*)d_in[4];
    const float* W2  = (const float*)d_in[5];
    // d_in[6] = b2 : cancels in BN, unused
    const float* g2  = (const float*)d_in[7];
    const float* be2 = (const float*)d_in[8];
    const float* W3  = (const float*)d_in[9];
    const float* b3  = (const float*)d_in[10];
    float* out = (float*)d_out;

    (void)d_ws; (void)ws_size; (void)in_sizes; (void)n_in; (void)out_size;

    hipFuncSetAttribute((const void*)loo_mlp_kernel,
                        hipFuncAttributeMaxDynamicSharedMemorySize, SMEM_BYTES);
    hipLaunchKernelGGL(loo_mlp_kernel, dim3(D_), dim3(1024), SMEM_BYTES, stream,
                       x, W1, g1, be1, W2, g2, be2, W3, b3, out);
}
